// CTransformerBlock_36876589203656
// MI455X (gfx1250) — matrix operand order repaired and checked by reading, not compiled, not run on hardware
//
#include <hip/hip_runtime.h>

#define B_ 4
#define N_ 4096
#define C_ 128
#define DP_ 32
#define K_ 24
#define EPSF 1e-5f
#define NQ_ 8

typedef __bf16 bf16;
typedef bf16 v16bf __attribute__((ext_vector_type(16)));
typedef float v8f __attribute__((ext_vector_type(8)));

// ---------------------------------------------------------------------------
// CDNA5 async direct-to-LDS copy (ASYNCcnt path, bypasses VGPRs).
// the LDS address is taken from the vdst VGPR; generic LDS pointers carry the LDS byte
// offset in their low 32 bits (ISA 10.2 aperture mapping), so we pass that.
// ---------------------------------------------------------------------------
__device__ __forceinline__ void async_load_b128(void* lds_dst, const void* gsrc) {
  unsigned off = (unsigned)(uintptr_t)lds_dst;
  asm volatile("global_load_async_to_lds_b128 %0, %1, off"
               :
               : "v"(off), "v"(gsrc)
               : "memory");
}
__device__ __forceinline__ void wait_async0() {
  asm volatile("s_wait_asynccnt 0" ::: "memory");
}

// ---------------------------------------------------------------------------
// WMMA tile loaders (wave32, 16-bit operands); this file's k order, same for A and B
// A 16x32 (MxK): lane L -> row M=L&15, K = (L>>4)*16 + 0..15 (contiguous)
// B 32x16 (KxN): lane L -> col N=L&15, K = (L>>4)*16 + 0..15 (contiguous)
// C/D 16x16 f32: VGPR v -> M = v + 8*(L>=16), N = L&15
// ---------------------------------------------------------------------------
__device__ __forceinline__ v16bf wmma_load_a(const bf16* A, int lda, int row_base,
                                             int k_base, int lane) {
  int r  = row_base + (lane & 15);
  int kg = (lane >> 4) * 16;
  const bf16* p0 = A + r * lda + k_base + kg;
  const bf16* p1 = p0 + 8;
  v16bf a;
#pragma unroll
  for (int i = 0; i < 8; ++i) { a[i] = p0[i]; a[8 + i] = p1[i]; }
  return a;
}

__device__ __forceinline__ v16bf wmma_load_b(const bf16* W, int ldw, int col_base,
                                             int k_base, int lane) {
  int c  = col_base + (lane & 15);
  int kh = (lane >> 4) * 16;
  const bf16* p = W + c * ldw + k_base + kh;
  v16bf b;
#pragma unroll
  for (int i = 0; i < 16; ++i) b[i] = p[i];
  return b;
}

// ---------------------------------------------------------------------------
__global__ void to_bf16_kernel(const float* __restrict__ src, bf16* __restrict__ dst, int n) {
  int i = blockIdx.x * 256 + threadIdx.x;
  if (i < n) dst[i] = (bf16)src[i];
}

// ---------------------------------------------------------------------------
// KNN: batch xyz (48KB) staged to LDS via async-to-LDS; one thread per query,
// register-resident sorted top-24 insertion (fully unrolled -> stays in VGPRs)
// grid: (N/256, B), block: 256
// ---------------------------------------------------------------------------
__global__ __launch_bounds__(256) void knn_kernel(const float* __restrict__ xyz,
                                                  int* __restrict__ knn_idx) {
  __shared__ float sx[N_ * 3];
  int b = blockIdx.y;
  const float* xb = xyz + (size_t)b * N_ * 3;
  for (int i = threadIdx.x; i < (N_ * 3) / 4; i += 256)
    async_load_b128(&sx[i * 4], &xb[i * 4]);
  wait_async0();
  __syncthreads();

  int n = blockIdx.x * 256 + threadIdx.x;
  float qx = sx[n * 3 + 0], qy = sx[n * 3 + 1], qz = sx[n * 3 + 2];

  float bd[K_];
  int   bi[K_];
#pragma unroll
  for (int j = 0; j < K_; ++j) { bd[j] = 3.0e38f; bi[j] = 0; }

  for (int m = 0; m < N_; ++m) {
    float dx = qx - sx[m * 3 + 0];
    float dy = qy - sx[m * 3 + 1];
    float dz = qz - sx[m * 3 + 2];
    float d  = dx * dx + dy * dy + dz * dz;
    if (d < bd[K_ - 1]) {
      int mi = m;
#pragma unroll
      for (int j = 0; j < K_; ++j) {
        bool sw = d < bd[j];
        float td = sw ? bd[j] : d;  int ti = sw ? bi[j] : mi;
        bd[j]    = sw ? d : bd[j];  bi[j]   = sw ? mi : bi[j];
        d = td; mi = ti;
      }
    }
  }
  int* out = knn_idx + ((size_t)b * N_ + n) * K_;
#pragma unroll
  for (int j = 0; j < K_; ++j) out[j] = bi[j];
}

// ---------------------------------------------------------------------------
// fc1: x = feat @ fc1_w^T + b   (K=32 -> single WMMA step)
// grid: (B*N/64), block: 256
// ---------------------------------------------------------------------------
__global__ __launch_bounds__(256) void fc1_wmma_kernel(const bf16* __restrict__ feat,
                                                       const bf16* __restrict__ w,
                                                       const float* __restrict__ bias,
                                                       float* __restrict__ xf,
                                                       bf16* __restrict__ xh) {
  __shared__ bf16 Wl[C_ * DP_];
  for (int i = threadIdx.x; i < (C_ * DP_) / 8; i += 256)
    async_load_b128(&Wl[i * 8], &w[i * 8]);
  wait_async0();
  __syncthreads();

  int lane = threadIdx.x & 31, wv = threadIdx.x >> 5;
  int colb = wv * 16;
  int col  = colb + (lane & 15);
  int mo   = (lane >> 4) * 8;
  float bc = bias[col];
  v16bf bt = wmma_load_b(Wl, DP_, colb, 0, lane);
  size_t mblk = (size_t)blockIdx.x * 64;

#pragma unroll
  for (int mt = 0; mt < 4; ++mt) {
    v8f acc = {0, 0, 0, 0, 0, 0, 0, 0};
    v16bf at = wmma_load_a(feat + (mblk + mt * 16) * DP_, DP_, 0, 0, lane);
    acc = __builtin_amdgcn_wmma_f32_16x16x32_bf16(false, at, false, bt, (short)0,
                                                  acc, false, false);
#pragma unroll
    for (int v = 0; v < 8; ++v) {
      size_t row = mblk + mt * 16 + mo + v;
      float val  = acc[v] + bc;
      xf[row * C_ + col] = val;
      xh[row * C_ + col] = (bf16)val;
    }
  }
}

// ---------------------------------------------------------------------------
// mean over N: mf[b][c] = mean_n x[b][n][c]      grid: B, block: 128
// ---------------------------------------------------------------------------
__global__ void mean_kernel(const float* __restrict__ x, float* __restrict__ mf) {
  int b = blockIdx.x, c = threadIdx.x;
  const float* p = x + (size_t)b * N_ * C_ + c;
  float s = 0.f;
  for (int n = 0; n < N_; ++n) s += p[(size_t)n * C_];
  mf[b * C_ + c] = s * (1.0f / N_);
}

// ---------------------------------------------------------------------------
// mixing matrix -> abs-normalized, stored transposed bf16: wcT[b][o][i]
// grid: (128, B), block: 128 (thread = o)
// ---------------------------------------------------------------------------
__global__ __launch_bounds__(128) void mixmat_kernel(const float* __restrict__ mf,
                                                     const float* __restrict__ W1,
                                                     const float* __restrict__ W2,
                                                     const float* __restrict__ Wc,
                                                     const float* __restrict__ bc,
                                                     bf16* __restrict__ wcT) {
  int i = blockIdx.x, b = blockIdx.y, o = threadIdx.x;
  __shared__ float smf[C_], sw1[C_], sw2[C_], red[C_];
  smf[o] = mf[b * C_ + o];
  sw1[o] = W1[i * C_ + o];
  sw2[o] = W2[i * C_ + o];
  __syncthreads();

  float mfi = smf[i];
  float acc = bc[o];
  const float* wcr = Wc + o * C_;
#pragma unroll 4
  for (int j = 0; j < C_; ++j)
    acc += (smf[j] * sw1[j] - mfi * sw2[j]) * wcr[j];

  red[o] = fabsf(acc);
  __syncthreads();
  for (int s = 64; s > 0; s >>= 1) {
    if (o < s) red[o] += red[o + s];
    __syncthreads();
  }
  float denom = red[0] + (float)C_ * EPSF;
  wcT[((size_t)b * C_ + o) * C_ + i] = (bf16)(acc / denom);
}

// ---------------------------------------------------------------------------
// batched GEMM: out[b,m,o] = sum_c A[b,m,c] * wc[b,c,o]  (+ optional residual)
// grid: (N/64, B), block: 256
// ---------------------------------------------------------------------------
__global__ __launch_bounds__(256) void bmm_wmma_kernel(const bf16* __restrict__ A,
                                                       const bf16* __restrict__ wcT,
                                                       const float* __restrict__ residual,
                                                       float* __restrict__ out,
                                                       bf16* __restrict__ outh) {
  __shared__ bf16 Wl[C_ * C_];
  int b = blockIdx.y;
  const bf16* wp = wcT + (size_t)b * C_ * C_;
  for (int i = threadIdx.x; i < (C_ * C_) / 8; i += 256)
    async_load_b128(&Wl[i * 8], &wp[i * 8]);
  wait_async0();
  __syncthreads();

  int lane = threadIdx.x & 31, wv = threadIdx.x >> 5;
  int colb = wv * 16;
  int col  = colb + (lane & 15);
  int mo   = (lane >> 4) * 8;
  size_t rowbase = (size_t)b * N_ + (size_t)blockIdx.x * 64;
  const bf16* Ab = A + rowbase * C_;

  v8f z = {0, 0, 0, 0, 0, 0, 0, 0};
  v8f acc[4] = {z, z, z, z};
#pragma unroll
  for (int kt = 0; kt < 4; ++kt) {
    v16bf bt = wmma_load_b(Wl, C_, colb, kt * 32, lane);
#pragma unroll
    for (int mt = 0; mt < 4; ++mt) {
      v16bf at = wmma_load_a(Ab, C_, mt * 16, kt * 32, lane);
      acc[mt] = __builtin_amdgcn_wmma_f32_16x16x32_bf16(false, at, false, bt, (short)0,
                                                        acc[mt], false, false);
    }
  }
#pragma unroll
  for (int mt = 0; mt < 4; ++mt)
#pragma unroll
    for (int v = 0; v < 8; ++v) {
      size_t gi = (rowbase + mt * 16 + mo + v) * C_ + col;
      float val = acc[mt][v];
      if (residual) val += residual[gi];
      out[gi] = val;
      if (outh) outh[gi] = (bf16)val;
    }
}

// ---------------------------------------------------------------------------
// in-block 32x128 GEMM (K=128) over LDS operands; W is [out][k] bf16 in LDS
// ---------------------------------------------------------------------------
__device__ __forceinline__ void gemm_32x128(const bf16* A, const bf16* W,
                                            const float* bias, bool relu,
                                            float* outF, bf16* outH, int tid) {
  int lane = tid & 31, wv = tid >> 5;
  int colb = wv * 16;
  int col  = colb + (lane & 15);
  int mo   = (lane >> 4) * 8;
  v8f acc0 = {0, 0, 0, 0, 0, 0, 0, 0};
  v8f acc1 = acc0;
#pragma unroll
  for (int kt = 0; kt < 4; ++kt) {
    v16bf bt = wmma_load_b(W, C_, colb, kt * 32, lane);
    v16bf a0 = wmma_load_a(A, C_, 0, kt * 32, lane);
    v16bf a1 = wmma_load_a(A, C_, 16, kt * 32, lane);
    acc0 = __builtin_amdgcn_wmma_f32_16x16x32_bf16(false, a0, false, bt, (short)0,
                                                   acc0, false, false);
    acc1 = __builtin_amdgcn_wmma_f32_16x16x32_bf16(false, a1, false, bt, (short)0,
                                                   acc1, false, false);
  }
  float bc = bias[col];
#pragma unroll
  for (int v = 0; v < 8; ++v) {
    float v0 = acc0[v] + bc, v1 = acc1[v] + bc;
    if (relu) { v0 = fmaxf(v0, 0.f); v1 = fmaxf(v1, 0.f); }
    int r0 = mo + v, r1 = 16 + mo + v;
    if (outF) { outF[r0 * C_ + col] = v0; outF[r1 * C_ + col] = v1; }
    if (outH) { outH[r0 * C_ + col] = (bf16)v0; outH[r1 * C_ + col] = (bf16)v1; }
  }
}

// ---------------------------------------------------------------------------
// fused attention: per query n, full K-pipeline in LDS.
// grid: B*(N/NQ_), block: 256. Dynamic LDS (see ATTN_SMEM).
// ---------------------------------------------------------------------------
__global__ __launch_bounds__(256) void attn_fused_kernel(
    const float* __restrict__ xyz, const int* __restrict__ knn_idx,
    const float* __restrict__ q, const float* __restrict__ kmat,
    const float* __restrict__ vmat,
    const float* __restrict__ delta_w1, const float* __restrict__ delta_b1,
    const float* __restrict__ delta_b2, const float* __restrict__ gamma_b1,
    const float* __restrict__ gamma_b2,
    const bf16* __restrict__ wd2, const bf16* __restrict__ wg1,
    const bf16* __restrict__ wg2,
    float* __restrict__ res1, bf16* __restrict__ res1h,
    float* __restrict__ attn_out) {
  extern __shared__ char smem[];
  bf16* sWd2 = (bf16*)smem;           // 128*128
  bf16* sWg1 = sWd2 + C_ * C_;
  bf16* sWg2 = sWg1 + C_ * C_;
  bf16* bufA = sWg2 + C_ * C_;        // 32*128
  bf16* bufB = bufA + 32 * C_;
  float* sPos = (float*)(bufB + 32 * C_);  // 32*128 f32
  float* sAtt = sPos + 32 * C_;
  float* sDen = sAtt + 32 * C_;       // 32
  float* sRel = sDen + 32;            // 96
  int*   sIdx = (int*)(sRel + 96);    // 32
  float* sB2  = (float*)(sIdx + 32);  // 128
  float* sG1  = sB2 + C_;
  float* sG2  = sG1 + C_;

  int tid = threadIdx.x;
  // async stage all three 128x128 bf16 weight matrices (96KB) into LDS
  for (int i = tid; i < (C_ * C_) / 8; i += 256) {
    async_load_b128(&sWd2[i * 8], &wd2[i * 8]);
    async_load_b128(&sWg1[i * 8], &wg1[i * 8]);
    async_load_b128(&sWg2[i * 8], &wg2[i * 8]);
  }
  if (tid < C_) { sB2[tid] = delta_b2[tid]; sG1[tid] = gamma_b1[tid]; sG2[tid] = gamma_b2[tid]; }
  wait_async0();
  __syncthreads();

  int b     = blockIdx.x / (N_ / NQ_);
  int chunk = blockIdx.x % (N_ / NQ_);

  for (int nq = 0; nq < NQ_; ++nq) {
    int n = chunk * NQ_ + nq;
    size_t rowq = (size_t)b * N_ + n;

    if (tid < K_) {
      int id = knn_idx[rowq * K_ + tid];
      sIdx[tid] = id;
      size_t rn = rowq * 3, rm = ((size_t)b * N_ + id) * 3;
      sRel[tid * 3 + 0] = xyz[rn + 0] - xyz[rm + 0];
      sRel[tid * 3 + 1] = xyz[rn + 1] - xyz[rm + 1];
      sRel[tid * 3 + 2] = xyz[rn + 2] - xyz[rm + 2];
    }
    __syncthreads();

    // h = relu(rel @ delta_w1^T + delta_b1) : 24x128 (rows 24..31 zero padded)
    for (int p = tid; p < 32 * C_; p += 256) {
      int k = p >> 7, c = p & 127;
      float v = 0.f;
      if (k < K_) {
        v = delta_b1[c] + sRel[k * 3 + 0] * delta_w1[c * 3 + 0]
                        + sRel[k * 3 + 1] * delta_w1[c * 3 + 1]
                        + sRel[k * 3 + 2] * delta_w1[c * 3 + 2];
        v = fmaxf(v, 0.f);
      }
      bufA[p] = (bf16)v;
    }
    __syncthreads();

    gemm_32x128(bufA, sWd2, sB2, false, sPos, nullptr, tid);  // pos_enc f32
    __syncthreads();

    // attn_in = q - k_gather + pos_enc
    for (int p = tid; p < 32 * C_; p += 256) {
      int k = p >> 7, c = p & 127;
      float v = 0.f;
      if (k < K_)
        v = q[rowq * C_ + c] - kmat[((size_t)b * N_ + sIdx[k]) * C_ + c] + sPos[p];
      bufA[p] = (bf16)v;
    }
    __syncthreads();

    gemm_32x128(bufA, sWg1, sG1, true, nullptr, bufB, tid);   // relu hidden
    __syncthreads();
    gemm_32x128(bufB, sWg2, sG2, false, sAtt, nullptr, tid);  // attn logits f32
    __syncthreads();

    if (tid < K_) {
      float s = 0.f;
#pragma unroll 4
      for (int c = 0; c < C_; ++c) s += fabsf(sAtt[tid * C_ + c]);
      sDen[tid] = s + (float)C_ * EPSF;
    }
    __syncthreads();

    // normalize + emit attn
    for (int p = tid; p < K_ * C_; p += 256) {
      int k = p >> 7;
      float a = sAtt[p] / sDen[k];
      sAtt[p] = a;
      attn_out[rowq * K_ * C_ + p] = a;
    }
    __syncthreads();

    // res1[c] = sum_k attn[k][c] * (v[k][c] + pos_enc[k][c])
    if (tid < C_) {
      int c = tid;
      float s = 0.f;
#pragma unroll
      for (int k = 0; k < K_; ++k)
        s += sAtt[k * C_ + c] *
             (vmat[((size_t)b * N_ + sIdx[k]) * C_ + c] + sPos[k * C_ + c]);
      res1[rowq * C_ + c]  = s;
      res1h[rowq * C_ + c] = (bf16)s;
    }
    __syncthreads();
  }
}

// dynamic LDS bytes for attn_fused_kernel
static const size_t ATTN_SMEM =
    (size_t)(3 * C_ * C_ + 2 * 32 * C_) * sizeof(bf16) +
    (size_t)(2 * 32 * C_) * sizeof(float) +
    (size_t)(32 + 96 + 32 + 3 * C_) * 4;

// ---------------------------------------------------------------------------
extern "C" void kernel_launch(void* const* d_in, const int* in_sizes, int n_in,
                              void* d_out, int out_size, void* d_ws, size_t ws_size,
                              hipStream_t stream) {
  const float* features = (const float*)d_in[0];
  const float* xyz      = (const float*)d_in[1];
  const float* fc1_w    = (const float*)d_in[2];
  const float* fc1_b    = (const float*)d_in[3];
  const float* q_W1 = (const float*)d_in[4],  *q_W2 = (const float*)d_in[5];
  const float* q_Wc = (const float*)d_in[6],  *q_bc = (const float*)d_in[7];
  const float* k_W1 = (const float*)d_in[8],  *k_W2 = (const float*)d_in[9];
  const float* k_Wc = (const float*)d_in[10], *k_bc = (const float*)d_in[11];
  const float* v_W1 = (const float*)d_in[12], *v_W2 = (const float*)d_in[13];
  const float* v_Wc = (const float*)d_in[14], *v_bc = (const float*)d_in[15];
  const float* f2_W1 = (const float*)d_in[16], *f2_W2 = (const float*)d_in[17];
  const float* f2_Wc = (const float*)d_in[18], *f2_bc = (const float*)d_in[19];
  const float* delta_w1 = (const float*)d_in[20];
  const float* delta_b1 = (const float*)d_in[21];
  const float* delta_w2 = (const float*)d_in[22];
  const float* delta_b2 = (const float*)d_in[23];
  const float* gamma_w1 = (const float*)d_in[24];
  const float* gamma_b1 = (const float*)d_in[25];
  const float* gamma_w2 = (const float*)d_in[26];
  const float* gamma_b2 = (const float*)d_in[27];

  // ----- workspace partition -----
  char* ws = (char*)d_ws;
  size_t off = 0;
  auto alloc = [&](size_t bytes) {
    size_t o = off;
    off = (off + bytes + 255) & ~(size_t)255;
    return (void*)(ws + o);
  };
  int*  idx     = (int*) alloc((size_t)B_ * N_ * K_ * 4);
  bf16* feat_h  = (bf16*)alloc((size_t)B_ * N_ * DP_ * 2);
  bf16* fc1w_h  = (bf16*)alloc((size_t)C_ * DP_ * 2);
  bf16* dw2_h   = (bf16*)alloc((size_t)C_ * C_ * 2);
  bf16* gw1_h   = (bf16*)alloc((size_t)C_ * C_ * 2);
  bf16* gw2_h   = (bf16*)alloc((size_t)C_ * C_ * 2);
  float* x_f    = (float*)alloc((size_t)B_ * N_ * C_ * 4);
  bf16*  x_h    = (bf16*) alloc((size_t)B_ * N_ * C_ * 2);
  float* mf     = (float*)alloc((size_t)B_ * C_ * 4);
  float* mf2    = (float*)alloc((size_t)B_ * C_ * 4);
  bf16* wcT_q   = (bf16*)alloc((size_t)B_ * C_ * C_ * 2);
  bf16* wcT_k   = (bf16*)alloc((size_t)B_ * C_ * C_ * 2);
  bf16* wcT_v   = (bf16*)alloc((size_t)B_ * C_ * C_ * 2);
  bf16* wcT_f2  = (bf16*)alloc((size_t)B_ * C_ * C_ * 2);
  float* q_f    = (float*)alloc((size_t)B_ * N_ * C_ * 4);
  float* k_f    = (float*)alloc((size_t)B_ * N_ * C_ * 4);
  float* v_f    = (float*)alloc((size_t)B_ * N_ * C_ * 4);
  float* res1_f = (float*)alloc((size_t)B_ * N_ * C_ * 4);
  bf16*  res1_h = (bf16*) alloc((size_t)B_ * N_ * C_ * 2);

  float* out_res  = (float*)d_out;
  float* out_attn = out_res + (size_t)B_ * N_ * C_;

  auto g1 = [](int n) { return dim3((unsigned)((n + 255) / 256)); };

  // precision conversions
  to_bf16_kernel<<<g1(B_ * N_ * DP_), 256, 0, stream>>>(features, feat_h, B_ * N_ * DP_);
  to_bf16_kernel<<<g1(C_ * DP_), 256, 0, stream>>>(fc1_w, fc1w_h, C_ * DP_);
  to_bf16_kernel<<<g1(C_ * C_), 256, 0, stream>>>(delta_w2, dw2_h, C_ * C_);
  to_bf16_kernel<<<g1(C_ * C_), 256, 0, stream>>>(gamma_w1, gw1_h, C_ * C_);
  to_bf16_kernel<<<g1(C_ * C_), 256, 0, stream>>>(gamma_w2, gw2_h, C_ * C_);

  // KNN
  knn_kernel<<<dim3(N_ / 256, B_), 256, 0, stream>>>(xyz, idx);

  // fc1
  fc1_wmma_kernel<<<dim3(B_ * N_ / 64), 256, 0, stream>>>(feat_h, fc1w_h, fc1_b, x_f, x_h);

  // q/k/v mixing matrices + batched GEMMs
  mean_kernel<<<dim3(B_), C_, 0, stream>>>(x_f, mf);
  mixmat_kernel<<<dim3(C_, B_), C_, 0, stream>>>(mf, q_W1, q_W2, q_Wc, q_bc, wcT_q);
  mixmat_kernel<<<dim3(C_, B_), C_, 0, stream>>>(mf, k_W1, k_W2, k_Wc, k_bc, wcT_k);
  mixmat_kernel<<<dim3(C_, B_), C_, 0, stream>>>(mf, v_W1, v_W2, v_Wc, v_bc, wcT_v);
  bmm_wmma_kernel<<<dim3(N_ / 64, B_), 256, 0, stream>>>(x_h, wcT_q, nullptr, q_f, nullptr);
  bmm_wmma_kernel<<<dim3(N_ / 64, B_), 256, 0, stream>>>(x_h, wcT_k, nullptr, k_f, nullptr);
  bmm_wmma_kernel<<<dim3(N_ / 64, B_), 256, 0, stream>>>(x_h, wcT_v, nullptr, v_f, nullptr);

  // fused attention pipeline
  attn_fused_kernel<<<dim3(B_ * (N_ / NQ_)), 256, ATTN_SMEM, stream>>>(
      xyz, idx, q_f, k_f, v_f, delta_w1, delta_b1, delta_b2, gamma_b1, gamma_b2,
      dw2_h, gw1_h, gw2_h, res1_f, res1_h, out_attn);

  // f2 mlinear + residual -> final res
  mean_kernel<<<dim3(B_), C_, 0, stream>>>(res1_f, mf2);
  mixmat_kernel<<<dim3(C_, B_), C_, 0, stream>>>(mf2, f2_W1, f2_W2, f2_Wc, f2_bc, wcT_f2);
  bmm_wmma_kernel<<<dim3(N_ / 64, B_), 256, 0, stream>>>(res1_h, wcT_f2, x_f, out_res, nullptr);
}
